// VQVAE_MAG_85950885528553
// MI455X (gfx1250) — matrix operand order repaired and checked by reading, not compiled, not run on hardware
//
#include <hip/hip_runtime.h>
#include <hip/hip_bf16.h>
#include <math.h>

#define DEVI __device__ __forceinline__

typedef __attribute__((ext_vector_type(16))) __bf16 v16bf;
typedef __attribute__((ext_vector_type(8)))  __bf16 v8bf;
typedef __attribute__((ext_vector_type(8)))  float  v8f;

static constexpr int Bsz  = 16384;   // batch rows
static constexpr int INsz = 128;     // input size
static constexpr int DENCc= 512;     // encoder dim
static constexpr int DOUTc= 256;     // encoder out dim (P*E_DIM)
static constexpr int NEc  = 1024;    // codebook entries
static constexpr int EDc  = 64;      // embedding dim
static constexpr int NVc  = Bsz * 4; // 65536 quantized vectors

// ---------------- bf16 helpers ----------------

DEVI __bf16 f2bf(float f) {
  unsigned u = __builtin_bit_cast(unsigned, f);
  unsigned r = u + 0x7FFFu + ((u >> 16) & 1u);     // round-to-nearest-even
  return __builtin_bit_cast(__bf16, (unsigned short)(r >> 16));
}

DEVI float bf2f(__bf16 b) {
  unsigned u = ((unsigned)__builtin_bit_cast(unsigned short, b)) << 16;
  return __builtin_bit_cast(float, u);
}

// A-matrix 16x32 bf16 fragment (ISA 7.12.2): lane m = lane&15, khalf = lane>>4,
// element e -> K = kb + (e&7) + 8*khalf + 16*(e>>3): two aligned 16B runs -> 2x b128.
DEVI v16bf load_a_bf(const __bf16* __restrict__ A, int row, int K, int kb, int half) {
  const __bf16* p = A + (size_t)row * K + kb + 8 * half;
  v8bf lo = *(const v8bf*)(p);
  v8bf hi = *(const v8bf*)(p + 16);
  return __builtin_shufflevector(lo, hi, 0, 1, 2, 3, 4, 5, 6, 7,
                                 8, 9, 10, 11, 12, 13, 14, 15);
}

// B fragment from pre-swizzled weights: frag (nt, ks) is 32 lanes x 16 bf16,
// lane-contiguous (1 KB per fragment) -> one 32B vector load per lane (2x b128).
DEVI v16bf load_b_sw(const __bf16* __restrict__ Wsw, int nt, int ks, int KS, int lane) {
  const __bf16* p = Wsw + ((((size_t)nt * KS + ks) * 32) + lane) * 16;
  return *(const v16bf*)p;
}

DEVI v8f wmma_bf16(v16bf a, v16bf b, v8f c) {
  return __builtin_amdgcn_wmma_f32_16x16x32_bf16(false, a, false, b, (short)0, c,
                                                 false, false);
}

// ---------------- weight swizzle (once per launch) ----------------
// W row-major f32 [N][K] -> bf16 fragments: out[((nt*KS+ks)*32+lane)*16 + e]
// = bf16(W[nt*16 + (lane&15)][ks*32 + 8*(lane>>4) + (e&7) + 16*(e>>3)])
__global__ void swizzle_w(const float* __restrict__ W, __bf16* __restrict__ Wsw,
                          int N, int K) {
  const int tid = blockIdx.x * blockDim.x + threadIdx.x;
  const int KS = K / 32;
  const int total = (N / 16) * KS * 32;
  if (tid >= total) return;
  const int lane = tid & 31;
  const int fs = tid >> 5;
  const int nt = fs / KS;
  const int ks = fs - nt * KS;
  const int n = nt * 16 + (lane & 15);
  const int k0 = ks * 32 + 8 * (lane >> 4);
  const float* src = W + (size_t)n * K + k0;
  __bf16* dst = Wsw + (size_t)tid * 16;
#pragma unroll
  for (int e = 0; e < 16; ++e) dst[e] = f2bf(src[e + (e & 8)]);
}

// ---------------- fused GEMM (+epilogue) ----------------
// out[row][col] = sum_k A[row][k] * W[col][k] + bias[col], then:
//   EPI 0: nothing more
//   EPI 1: out = resid + relu(LN(h)*g + be)                 (resblock)
//   EPI 2: out = (resid + relu(LN(h)*g + be)) * mul         (resblock * av)
//   EPI 3: out = sc[col]*h + sh[col]                        (final de-normalize)
// Block: 256 threads = 8 waves; 64 rows x N cols per block.
// Wave (rg = wave>>2, cg = wave&3): 32 rows (2 M-tiles) x N/4 cols (NT N-tiles);
// each B fragment is reused by two WMMAs -> 1.25 vmem per WMMA.

template <int N, int K, int EPI>
__launch_bounds__(256)
__global__ void gemm_wmma(const __bf16* __restrict__ A,
                          const __bf16* __restrict__ Wsw,
                          const float* __restrict__ bias,
                          const __bf16* __restrict__ resid,
                          const float* __restrict__ gamma,
                          const float* __restrict__ beta,
                          const __bf16* __restrict__ mul,
                          const float* __restrict__ sc,
                          const float* __restrict__ sh,
                          __bf16* __restrict__ outb,
                          float* __restrict__ outf) {
  constexpr int NT = N / 64;
  constexpr int KS = K / 32;
  const int wave = threadIdx.x >> 5;
  const int lane = threadIdx.x & 31;
  const int rg   = wave >> 2;
  const int cg   = wave & 3;
  const int half = lane >> 4;
  const int ln15 = lane & 15;
  const int blockrow = blockIdx.x * 64;
  const int rowb0 = blockrow + rg * 32;      // wave's 32-row strip
  const int colbase = cg * (N / 4);
  const int ntbase  = cg * NT;

  v8f acc[2][NT];
#pragma unroll
  for (int mt = 0; mt < 2; ++mt)
#pragma unroll
    for (int t = 0; t < NT; ++t)
#pragma unroll
      for (int r = 0; r < 8; ++r) acc[mt][t][r] = 0.0f;

#pragma unroll 2
  for (int ks = 0; ks < KS; ++ks) {
    if (ks + 1 < KS) {  // gfx1250 global_prefetch_b8 of the next k-slab
      __builtin_prefetch(A + (size_t)(rowb0 + ln15) * K + ks * 32 + 32, 0, 0);
      __builtin_prefetch(Wsw + (((size_t)ntbase * KS + ks + 1) * 32 + lane) * 16, 0, 0);
    }
    const v16bf a0 = load_a_bf(A, rowb0 + ln15,      K, ks * 32, half);
    const v16bf a1 = load_a_bf(A, rowb0 + 16 + ln15, K, ks * 32, half);
#pragma unroll
    for (int t = 0; t < NT; ++t) {
      const v16bf bf_ = load_b_sw(Wsw, ntbase + t, ks, KS, lane);
      acc[0][t] = wmma_bf16(a0, bf_, acc[0][t]);
      acc[1][t] = wmma_bf16(a1, bf_, acc[1][t]);
    }
  }

  // bias added in place (C/D layout: elem r -> row mt*16 + 8*half + r, col t*16+ln15)
#pragma unroll
  for (int t = 0; t < NT; ++t) {
    const float bv = bias[colbase + t * 16 + ln15];
#pragma unroll
    for (int mt = 0; mt < 2; ++mt)
#pragma unroll
      for (int r = 0; r < 8; ++r) acc[mt][t][r] += bv;
  }

  __shared__ float red1[64][4];
  __shared__ float red2[64][4];

  if constexpr (EPI == 1 || EPI == 2) {
    // row-wise LayerNorm over full N: wave shuffle reduce + LDS cross-wave combine
#pragma unroll
    for (int mt = 0; mt < 2; ++mt) {
      float s1[8], s2[8];
#pragma unroll
      for (int r = 0; r < 8; ++r) {
        s1[r] = 0.0f; s2[r] = 0.0f;
#pragma unroll
        for (int t = 0; t < NT; ++t) {
          s1[r] += acc[mt][t][r];
          s2[r] += acc[mt][t][r] * acc[mt][t][r];
        }
#pragma unroll
        for (int off = 1; off < 16; off <<= 1) {
          s1[r] += __shfl_xor(s1[r], off, 32);
          s2[r] += __shfl_xor(s2[r], off, 32);
        }
      }
      if (ln15 == 0) {
        const int mb = rg * 32 + mt * 16 + 8 * half;
#pragma unroll
        for (int r = 0; r < 8; ++r) {
          red1[mb + r][cg] = s1[r];
          red2[mb + r][cg] = s2[r];
        }
      }
    }
    __syncthreads();
#pragma unroll
    for (int mt = 0; mt < 2; ++mt) {
#pragma unroll
      for (int r = 0; r < 8; ++r) {
        const int lm  = rg * 32 + mt * 16 + 8 * half + r;
        const int row = blockrow + lm;
        const float sum = red1[lm][0] + red1[lm][1] + red1[lm][2] + red1[lm][3];
        const float sq  = red2[lm][0] + red2[lm][1] + red2[lm][2] + red2[lm][3];
        const float mean = sum * (1.0f / N);
        const float var  = sq * (1.0f / N) - mean * mean;
        const float inv  = rsqrtf(var + 1e-5f);
#pragma unroll
        for (int t = 0; t < NT; ++t) {
          const int col = colbase + t * 16 + ln15;
          float v = (acc[mt][t][r] - mean) * inv * gamma[col] + beta[col];
          v = fmaxf(v, 0.0f);
          float o = bf2f(resid[(size_t)row * N + col]) + v;
          if constexpr (EPI == 2) o *= bf2f(mul[(size_t)row * N + col]);
          outb[(size_t)row * N + col] = f2bf(o);
          if (outf) outf[(size_t)row * N + col] = o;
        }
      }
    }
  } else {
#pragma unroll
    for (int mt = 0; mt < 2; ++mt) {
#pragma unroll
      for (int r = 0; r < 8; ++r) {
        const int row = rowb0 + mt * 16 + 8 * half + r;
#pragma unroll
        for (int t = 0; t < NT; ++t) {
          const int col = colbase + t * 16 + ln15;
          float v = acc[mt][t][r];
          if constexpr (EPI == 3) v = sc[col] * v + sh[col];
          if (outb) outb[(size_t)row * N + col] = f2bf(v);
          if (outf) outf[(size_t)row * N + col] = v;
        }
      }
    }
  }
}

// ---------------- VQ nearest-codebook via WMMA ----------------
// argmin_n ||z - c_n||^2 = argmax_n (z.c_n - 0.5*||c_n||^2); seed WMMA C with the
// -0.5||c||^2 term. Each wave: 16 rows, full 1024-entry sweep in 64 N-tiles.
__launch_bounds__(256)
__global__ void vq_argmin(const __bf16* __restrict__ zeb,   // [NV][64] bf16
                          const __bf16* __restrict__ cbsw,  // swizzled codebook
                          const float* __restrict__ cn2,    // [1024]
                          int* __restrict__ idx_i,
                          float* __restrict__ idx_f) {
  const int wave = threadIdx.x >> 5;
  const int lane = threadIdx.x & 31;
  const int half = lane >> 4;
  const int ln15 = lane & 15;
  const int rowbase = blockIdx.x * 128 + wave * 16;

  const v16bf a0 = load_a_bf(zeb, rowbase + ln15, EDc, 0, half);
  const v16bf a1 = load_a_bf(zeb, rowbase + ln15, EDc, 32, half);

  float best[8];
  int   bidx[8];
#pragma unroll
  for (int r = 0; r < 8; ++r) { best[r] = -3.0e38f; bidx[r] = 0; }

#pragma unroll 4
  for (int nt = 0; nt < NEc / 16; ++nt) {
    const int col = nt * 16 + ln15;
    const float cinit = -0.5f * cn2[col];
    v8f acc;
#pragma unroll
    for (int r = 0; r < 8; ++r) acc[r] = cinit;
    const v16bf b0 = load_b_sw(cbsw, nt, 0, 2, lane);
    const v16bf b1 = load_b_sw(cbsw, nt, 1, 2, lane);
    acc = wmma_bf16(a0, b0, acc);
    acc = wmma_bf16(a1, b1, acc);
#pragma unroll
    for (int r = 0; r < 8; ++r)
      if (acc[r] > best[r]) { best[r] = acc[r]; bidx[r] = col; }
  }
  // argmax across the 16 lanes holding this row (ties -> smaller index, like argmin)
#pragma unroll
  for (int off = 1; off < 16; off <<= 1) {
#pragma unroll
    for (int r = 0; r < 8; ++r) {
      const float ob = __shfl_xor(best[r], off, 32);
      const int   oi = __shfl_xor(bidx[r], off, 32);
      if (ob > best[r] || (ob == best[r] && oi < bidx[r])) { best[r] = ob; bidx[r] = oi; }
    }
  }
  if (ln15 == 0) {
#pragma unroll
    for (int r = 0; r < 8; ++r) {
      const int row = rowbase + 8 * half + r;
      idx_i[row] = bidx[r];
      idx_f[row] = (float)bidx[r];
    }
  }
}

// ---------------- small glue kernels ----------------

__global__ void prep_inputs(const float* __restrict__ x, const float* __restrict__ a,
                            const float* __restrict__ scale,
                            const float* __restrict__ shift,
                            __bf16* __restrict__ xsb, __bf16* __restrict__ ab) {
  const int i = blockIdx.x * blockDim.x + threadIdx.x;  // Bsz*INsz threads
  const int c = i & (INsz - 1);
  xsb[i] = f2bf((x[i] - shift[c]) / scale[c]);
  ab[i]  = f2bf(a[i]);
}

__global__ void cb_norms(const float* __restrict__ cb, float* __restrict__ cn2) {
  const int n = blockIdx.x * blockDim.x + threadIdx.x;
  if (n < NEc) {
    float s = 0.0f;
#pragma unroll 8
    for (int j = 0; j < EDc; ++j) { const float v = cb[n * EDc + j]; s += v * v; }
    cn2[n] = s;
  }
}

__global__ void zero_stats(float* __restrict__ counts, float* __restrict__ lacc) {
  counts[threadIdx.x] = 0.0f;
  if (threadIdx.x == 0) lacc[0] = 0.0f;
}

__global__ void gather_loss(const float* __restrict__ ze, const float* __restrict__ cb,
                            const int* __restrict__ idx_i,
                            __bf16* __restrict__ zq_bf, float* __restrict__ zq_out,
                            float* __restrict__ counts, float* __restrict__ lacc) {
  const int row = blockIdx.x * blockDim.x + threadIdx.x;  // NV threads
  const int id  = idx_i[row];
  float s = 0.0f;
#pragma unroll 8
  for (int j = 0; j < EDc; ++j) {
    const float q = cb[id * EDc + j];
    const float e = ze[(size_t)row * EDc + j];
    zq_bf[(size_t)row * EDc + j]  = f2bf(q);
    zq_out[(size_t)row * EDc + j] = q;   // z_q_st == z_q in forward value
    const float d = q - e;
    s += d * d;
  }
  atomicAdd(&counts[id], 1.0f);
  __shared__ float red[256];
  red[threadIdx.x] = s;
  __syncthreads();
  for (int st = 128; st > 0; st >>= 1) {
    if (threadIdx.x < st) red[threadIdx.x] += red[threadIdx.x + st];
    __syncthreads();
  }
  if (threadIdx.x == 0) atomicAdd(lacc, red[0]);
}

__global__ void finalize_stats(const float* __restrict__ counts,
                               const float* __restrict__ lacc,
                               float* __restrict__ out_loss,
                               float* __restrict__ out_perp) {
  __shared__ float red[NEc];
  const int t = threadIdx.x;
  const float em = counts[t] * (1.0f / (float)NVc);
  red[t] = -em * logf(em + 1e-10f);
  __syncthreads();
  for (int st = NEc / 2; st > 0; st >>= 1) {
    if (t < st) red[t] += red[t + st];
    __syncthreads();
  }
  if (t == 0) {
    out_perp[0] = expf(red[0]);
    // loss = mse + BETA*mse (stop_gradients don't change forward value)
    out_loss[0] = 1.25f * lacc[0] / ((float)NVc * (float)EDc);
  }
}

// ---------------- host orchestration ----------------

extern "C" void kernel_launch(void* const* d_in, const int* in_sizes, int n_in,
                              void* d_out, int out_size, void* d_ws, size_t ws_size,
                              hipStream_t stream) {
  (void)in_sizes; (void)n_in; (void)out_size; (void)ws_size;

  const float* x          = (const float*)d_in[0];
  const float* a          = (const float*)d_in[1];
  const float* scale      = (const float*)d_in[2];
  const float* shift      = (const float*)d_in[3];
  const float* enc_win_w  = (const float*)d_in[4];
  const float* enc_win_b  = (const float*)d_in[5];
  const float* enc_ain_w  = (const float*)d_in[6];
  const float* enc_ain_b  = (const float*)d_in[7];
  const float* enc_wres_w = (const float*)d_in[8];
  const float* enc_wres_b = (const float*)d_in[9];
  const float* enc_wres_g = (const float*)d_in[10];
  const float* enc_wres_be= (const float*)d_in[11];
  const float* enc_ares_w = (const float*)d_in[12];
  const float* enc_ares_b = (const float*)d_in[13];
  const float* enc_ares_g = (const float*)d_in[14];
  const float* enc_ares_be= (const float*)d_in[15];
  const float* enc_out_w  = (const float*)d_in[16];
  const float* enc_out_b  = (const float*)d_in[17];
  const float* dec1_w     = (const float*)d_in[18];
  const float* dec1_b     = (const float*)d_in[19];
  const float* dec_res_w  = (const float*)d_in[20];
  const float* dec_res_b  = (const float*)d_in[21];
  const float* dec_res_g  = (const float*)d_in[22];
  const float* dec_res_be = (const float*)d_in[23];
  const float* dec2_w     = (const float*)d_in[24];
  const float* dec2_b     = (const float*)d_in[25];
  const float* codebook   = (const float*)d_in[26];

  // d_out layout (floats, reference tuple order): loss, x_hat, perplexity, z_q_st, idx
  float* out      = (float*)d_out;
  float* out_loss = out;
  float* out_xhat = out + 1;
  float* out_perp = out + 1 + (size_t)Bsz * INsz;
  float* out_zq   = out + 2 + (size_t)Bsz * INsz;
  float* out_idx  = out + 2 + (size_t)Bsz * INsz + (size_t)Bsz * DOUTc;

  // workspace carve-up
  char* wsp = (char*)d_ws;
  auto carve = [&](size_t bytes) {
    char* p = wsp;
    wsp += (bytes + 255) & ~(size_t)255;
    return p;
  };
  // bf16 activations
  __bf16* xs_bf = (__bf16*)carve((size_t)Bsz * INsz  * 2);
  __bf16* a_bf  = (__bf16*)carve((size_t)Bsz * INsz  * 2);
  __bf16* av_a  = (__bf16*)carve((size_t)Bsz * DENCc * 2);
  __bf16* av_b  = (__bf16*)carve((size_t)Bsz * DENCc * 2);
  __bf16* h_a   = (__bf16*)carve((size_t)Bsz * DENCc * 2);
  __bf16* h_b   = (__bf16*)carve((size_t)Bsz * DENCc * 2);
  __bf16* ze_bf = (__bf16*)carve((size_t)Bsz * DOUTc * 2);
  __bf16* zq_bf = (__bf16*)carve((size_t)Bsz * DOUTc * 2);
  __bf16* hd_a  = (__bf16*)carve((size_t)Bsz * DENCc * 2);
  __bf16* hd_b  = (__bf16*)carve((size_t)Bsz * DENCc * 2);
  float*  ze    = (float*)carve((size_t)Bsz * DOUTc * 4);
  // swizzled bf16 weights
  __bf16* sw_win  = (__bf16*)carve((size_t)DENCc * INsz  * 2);
  __bf16* sw_ain  = (__bf16*)carve((size_t)DENCc * INsz  * 2);
  __bf16* sw_wres = (__bf16*)carve((size_t)DENCc * DENCc * 2);
  __bf16* sw_ares = (__bf16*)carve((size_t)DENCc * DENCc * 2);
  __bf16* sw_eout = (__bf16*)carve((size_t)DOUTc * DENCc * 2);
  __bf16* sw_dec1 = (__bf16*)carve((size_t)DENCc * DOUTc * 2);
  __bf16* sw_dres = (__bf16*)carve((size_t)DENCc * DENCc * 2);
  __bf16* sw_dec2 = (__bf16*)carve((size_t)INsz  * DENCc * 2);
  __bf16* sw_cb   = (__bf16*)carve((size_t)NEc   * EDc   * 2);
  // VQ stats
  float* cn2  = (float*)carve((size_t)NEc * 4);
  int*   idxi = (int*)  carve((size_t)NVc * 4);
  float* cnts = (float*)carve((size_t)NEc * 4);
  float* lacc = (float*)carve(256);

  const dim3 blk(256);
  const dim3 gG(Bsz / 64);   // 256 blocks per GEMM (64 rows per block)
  auto swgrid = [](int N, int K) { return dim3(((N / 16) * (K / 32) * 32 + 255) / 256); };

  // one-time-per-launch conversions / swizzles
  prep_inputs<<<dim3((Bsz * INsz) / 256), blk, 0, stream>>>(x, a, scale, shift, xs_bf, a_bf);
  zero_stats<<<dim3(1), dim3(NEc), 0, stream>>>(cnts, lacc);
  cb_norms<<<dim3(NEc / 256), blk, 0, stream>>>(codebook, cn2);
  swizzle_w<<<swgrid(DENCc, INsz),  blk, 0, stream>>>(enc_win_w,  sw_win,  DENCc, INsz);
  swizzle_w<<<swgrid(DENCc, INsz),  blk, 0, stream>>>(enc_ain_w,  sw_ain,  DENCc, INsz);
  swizzle_w<<<swgrid(DENCc, DENCc), blk, 0, stream>>>(enc_wres_w, sw_wres, DENCc, DENCc);
  swizzle_w<<<swgrid(DENCc, DENCc), blk, 0, stream>>>(enc_ares_w, sw_ares, DENCc, DENCc);
  swizzle_w<<<swgrid(DOUTc, DENCc), blk, 0, stream>>>(enc_out_w,  sw_eout, DOUTc, DENCc);
  swizzle_w<<<swgrid(DENCc, DOUTc), blk, 0, stream>>>(dec1_w,     sw_dec1, DENCc, DOUTc);
  swizzle_w<<<swgrid(DENCc, DENCc), blk, 0, stream>>>(dec_res_w,  sw_dres, DENCc, DENCc);
  swizzle_w<<<swgrid(INsz,  DENCc), blk, 0, stream>>>(dec2_w,     sw_dec2, INsz,  DENCc);
  swizzle_w<<<swgrid(NEc,   EDc),   blk, 0, stream>>>(codebook,   sw_cb,   NEc,   EDc);

  // encoder input projections
  gemm_wmma<DENCc, INsz, 0><<<gG, blk, 0, stream>>>(
      a_bf, sw_ain, enc_ain_b, nullptr, nullptr, nullptr, nullptr, nullptr, nullptr,
      av_a, nullptr);
  gemm_wmma<DENCc, INsz, 0><<<gG, blk, 0, stream>>>(
      xs_bf, sw_win, enc_win_b, nullptr, nullptr, nullptr, nullptr, nullptr, nullptr,
      h_a, nullptr);

  // 3 shared-weight resblock pairs, h path modulated by av path
  __bf16 *avi = av_a, *avo = av_b, *hi = h_a, *ho = h_b;
  for (int i = 0; i < 3; ++i) {
    gemm_wmma<DENCc, DENCc, 1><<<gG, blk, 0, stream>>>(
        avi, sw_ares, enc_ares_b, avi, enc_ares_g, enc_ares_be, nullptr, nullptr,
        nullptr, avo, nullptr);
    gemm_wmma<DENCc, DENCc, 2><<<gG, blk, 0, stream>>>(
        hi, sw_wres, enc_wres_b, hi, enc_wres_g, enc_wres_be, avo, nullptr, nullptr,
        ho, nullptr);
    __bf16* t = avi; avi = avo; avo = t;
    t = hi; hi = ho; ho = t;
  }

  // bottleneck: z_e in both bf16 (VQ matmul) and f32 (loss/gather)
  gemm_wmma<DOUTc, DENCc, 0><<<gG, blk, 0, stream>>>(
      hi, sw_eout, enc_out_b, nullptr, nullptr, nullptr, nullptr, nullptr, nullptr,
      ze_bf, ze);

  // vector quantization
  vq_argmin<<<dim3(NVc / 128), blk, 0, stream>>>(ze_bf, sw_cb, cn2, idxi, out_idx);
  gather_loss<<<dim3(NVc / 256), blk, 0, stream>>>(ze, codebook, idxi, zq_bf, out_zq,
                                                   cnts, lacc);
  finalize_stats<<<dim3(1), dim3(NEc), 0, stream>>>(cnts, lacc, out_loss, out_perp);

  // decoder
  gemm_wmma<DENCc, DOUTc, 0><<<gG, blk, 0, stream>>>(
      zq_bf, sw_dec1, dec1_b, nullptr, nullptr, nullptr, nullptr, nullptr, nullptr,
      hd_a, nullptr);
  __bf16 *di = hd_a, *dd = hd_b;
  for (int i = 0; i < 3; ++i) {
    gemm_wmma<DENCc, DENCc, 1><<<gG, blk, 0, stream>>>(
        di, sw_dres, dec_res_b, di, dec_res_g, dec_res_be, nullptr, nullptr, nullptr,
        dd, nullptr);
    __bf16* t = di; di = dd; dd = t;
  }
  gemm_wmma<INsz, DENCc, 3><<<gG, blk, 0, stream>>>(
      di, sw_dec2, dec2_b, nullptr, nullptr, nullptr, nullptr, scale, shift,
      nullptr, out_xhat);
}
